// PointSIS_Feature_Extractor_30545807409340
// MI455X (gfx1250) — matrix operand order repaired and checked by reading, not compiled, not run on hardware
//
#include <hip/hip_runtime.h>
#include <hip/hip_bf16.h>
#include <math.h>

#define NPTS 8192
#define GG   4096
#define BS   2
#define DD   256
#define DI   512
#define NQ   100
#define NQR  200   /* BS*NQ valid rows */
#define NQP  256   /* padded rows for guard-free GEMM */

typedef __attribute__((ext_vector_type(16))) __bf16 v16bf;
typedef __attribute__((ext_vector_type(8)))  float  v8f;

// pack two fp32 into two bf16 (truncation) with a single v_perm_b32:
// result = { hi[31:16], lo[31:16] }
__device__ __forceinline__ unsigned pack2bf(float lo, float hi){
  return __builtin_amdgcn_perm(__float_as_uint(hi), __float_as_uint(lo), 0x07060302u);
}
__device__ __forceinline__ float gelu_f(float x){
  float t = tanhf(0.7978845608028654f*(x + 0.044715f*x*x*x));
  return 0.5f*x*(1.f+t);
}
__device__ __forceinline__ float silu_f(float x){ return x/(1.f+__expf(-x)); }
__device__ __forceinline__ float softplus_f(float x){ return (x>20.f)?x:log1pf(__expf(x)); }

// ---------------------------------------------------------------------------
// Guard-free WMMA GEMM: C[M,N] = act(A[M,K](lda) @ B[K,N](ldb) + bias)
// Preconditions (arranged by padding): M%128==0, N%64==0, K%32==0,
// A rows 16B-aligned (lda%8==0). 256 threads = 8 waves, each wave owns a
// 32x32 tile (4x v_wmma_f32_16x16x32_bf16 per 32-K step).
// act: 0=none 1=gelu 2=silu 3=softplus
// ---------------------------------------------------------------------------
__global__ __launch_bounds__(256)
void gemm_kernel(const float* __restrict__ A, int lda,
                 const float* __restrict__ B, int ldb,
                 const float* __restrict__ bias,
                 float* __restrict__ C, int ldc,
                 int K, int act)
{
  const int lane = threadIdx.x & 31;
  const int wave = threadIdx.x >> 5;
  const int wm = wave & 3;               // 4 waves along M
  const int wn = wave >> 2;              // 2 waves along N
  const int tileM = blockIdx.y * 128 + wm * 32;
  const int tileN = blockIdx.x * 64  + wn * 32;
  const int hl  = lane >> 4;             // half-wave select
  const int l16 = lane & 15;
  // A fragment layout (contiguous K, matching B):
  //  lanes 0-15:  elems 0-15 -> K=k0+0..15
  //  lanes 16-31: elems 0-15 -> K=k0+16..31
  const int akb = hl ? 16 : 0;
  // B fragment (32x16): lanes 0-15 K=k0..k0+15, lanes 16-31 K=k0+16..k0+31
  const int bkb = hl ? 16 : 0;

  const float* arow0 = A + (size_t)(tileM + l16) * lda + akb;
  const float* arow1 = arow0 + (size_t)16 * lda;
  const float* bp0   = B + (size_t)bkb * ldb + (tileN + l16);
  const float* bp1   = bp0 + 16;

  union BfVec { v16bf v; unsigned u[8]; };
  v8f acc00 = {}, acc01 = {}, acc10 = {}, acc11 = {};
  for (int k0 = 0; k0 < K; k0 += 32) {
    BfVec a0, a1, b0, b1;
    {
      float4 u0 = *(const float4*)(arow0 + k0);
      float4 u1 = *(const float4*)(arow0 + k0 + 4);
      float4 u2 = *(const float4*)(arow0 + k0 + 8);
      float4 u3 = *(const float4*)(arow0 + k0 + 12);
      a0.u[0]=pack2bf(u0.x,u0.y); a0.u[1]=pack2bf(u0.z,u0.w);
      a0.u[2]=pack2bf(u1.x,u1.y); a0.u[3]=pack2bf(u1.z,u1.w);
      a0.u[4]=pack2bf(u2.x,u2.y); a0.u[5]=pack2bf(u2.z,u2.w);
      a0.u[6]=pack2bf(u3.x,u3.y); a0.u[7]=pack2bf(u3.z,u3.w);
    }
    {
      float4 u0 = *(const float4*)(arow1 + k0);
      float4 u1 = *(const float4*)(arow1 + k0 + 4);
      float4 u2 = *(const float4*)(arow1 + k0 + 8);
      float4 u3 = *(const float4*)(arow1 + k0 + 12);
      a1.u[0]=pack2bf(u0.x,u0.y); a1.u[1]=pack2bf(u0.z,u0.w);
      a1.u[2]=pack2bf(u1.x,u1.y); a1.u[3]=pack2bf(u1.z,u1.w);
      a1.u[4]=pack2bf(u2.x,u2.y); a1.u[5]=pack2bf(u2.z,u2.w);
      a1.u[6]=pack2bf(u3.x,u3.y); a1.u[7]=pack2bf(u3.z,u3.w);
    }
#pragma unroll
    for (int j = 0; j < 8; ++j) {
      size_t o = (size_t)(k0 + 2*j) * ldb;
      float p0 = bp0[o], p1 = bp0[o + ldb];
      float q0 = bp1[o], q1 = bp1[o + ldb];
      b0.u[j] = pack2bf(p0, p1);
      b1.u[j] = pack2bf(q0, q1);
    }
    acc00 = __builtin_amdgcn_wmma_f32_16x16x32_bf16(false, a0.v, false, b0.v, (short)0, acc00, false, false);
    acc01 = __builtin_amdgcn_wmma_f32_16x16x32_bf16(false, a0.v, false, b1.v, (short)0, acc01, false, false);
    acc10 = __builtin_amdgcn_wmma_f32_16x16x32_bf16(false, a1.v, false, b0.v, (short)0, acc10, false, false);
    acc11 = __builtin_amdgcn_wmma_f32_16x16x32_bf16(false, a1.v, false, b1.v, (short)0, acc11, false, false);
  }
  // D layout: VGPR r -> M = rbase+r (lanes 0-15) / rbase=8 (lanes 16-31)
  const int rbase = hl * 8;
  const int col0 = tileN + l16, col1 = col0 + 16;
  float bv0 = bias ? bias[col0] : 0.f;
  float bv1 = bias ? bias[col1] : 0.f;
#pragma unroll
  for (int r = 0; r < 8; ++r) {
    int row0 = tileM + rbase + r, row1 = row0 + 16;
    float v00 = acc00[r] + bv0, v01 = acc01[r] + bv1;
    float v10 = acc10[r] + bv0, v11 = acc11[r] + bv1;
    if (act == 1) { v00=gelu_f(v00); v01=gelu_f(v01); v10=gelu_f(v10); v11=gelu_f(v11); }
    else if (act == 2) { v00=silu_f(v00); v01=silu_f(v01); v10=silu_f(v10); v11=silu_f(v11); }
    else if (act == 3) { v00=softplus_f(v00); v01=softplus_f(v01); v10=softplus_f(v10); v11=softplus_f(v11); }
    C[(size_t)row0*ldc + col0] = v00;
    C[(size_t)row0*ldc + col1] = v01;
    C[(size_t)row1*ldc + col0] = v10;
    C[(size_t)row1*ldc + col1] = v11;
  }
}

// ---------------------------------------------------------------------------
// LayerNorm over D=256: out = LN(a (+ b)) * w + bias ; one row per block
// ---------------------------------------------------------------------------
__global__ __launch_bounds__(256)
void ln_kernel(const float* __restrict__ a, const float* __restrict__ b,
               const float* __restrict__ w, const float* __restrict__ bias,
               float* __restrict__ out)
{
  int row = blockIdx.x, tid = threadIdx.x;
  size_t idx = (size_t)row*DD + tid;
  float x = a[idx];
  if (b) x += b[idx];
  __shared__ float red[256];
  red[tid] = x; __syncthreads();
  for (int off = 128; off; off >>= 1) { if (tid < off) red[tid] += red[tid+off]; __syncthreads(); }
  float mu = red[0] * (1.f/DD); __syncthreads();
  float d0 = x - mu;
  red[tid] = d0*d0; __syncthreads();
  for (int off = 128; off; off >>= 1) { if (tid < off) red[tid] += red[tid+off]; __syncthreads(); }
  float var = red[0] * (1.f/DD);
  out[idx] = d0 * rsqrtf(var + 1e-5f) * w[tid] + bias[tid];
}

// ---------------------------------------------------------------------------
// Multi-head attention core (per (q,head,batch) block). HEAD_DIM=32, 8 heads.
// ---------------------------------------------------------------------------
__global__ __launch_bounds__(256)
void attn_kernel(const float* __restrict__ Q, const float* __restrict__ Kp,
                 const float* __restrict__ Vp, float* __restrict__ O,
                 int nq, int nk)
{
  int qi = blockIdx.x, h = blockIdx.y, b = blockIdx.z;
  int tid = threadIdx.x;
  __shared__ float sq[32];
  __shared__ float red[256];
  __shared__ float sO[32];
  if (tid < 32) { sq[tid] = Q[((size_t)(b*nq+qi))*DD + h*32 + tid]; sO[tid] = 0.f; }
  __syncthreads();
  float sc[16];
  float lmax = -3.0e38f;
  int cnt = 0;
  for (int m = tid; m < nk; m += 256) {
    const float* kp = Kp + ((size_t)(b*nk+m))*DD + h*32;
    float s = 0.f;
#pragma unroll
    for (int d2 = 0; d2 < 32; ++d2) s += sq[d2]*kp[d2];
    sc[cnt] = s * 0.17677669529663687f;           // 1/sqrt(32)
    lmax = fmaxf(lmax, sc[cnt]); ++cnt;
  }
  red[tid] = lmax; __syncthreads();
  for (int off = 128; off; off >>= 1) { if (tid < off) red[tid] = fmaxf(red[tid], red[tid+off]); __syncthreads(); }
  float gmax = red[0]; __syncthreads();
  float acc[32];
#pragma unroll
  for (int d2 = 0; d2 < 32; ++d2) acc[d2] = 0.f;
  float lsum = 0.f; cnt = 0;
  for (int m = tid; m < nk; m += 256) {
    float p = __expf(sc[cnt++] - gmax);
    lsum += p;
    const float* vp = Vp + ((size_t)(b*nk+m))*DD + h*32;
#pragma unroll
    for (int d2 = 0; d2 < 32; ++d2) acc[d2] += p*vp[d2];
  }
  red[tid] = lsum; __syncthreads();
  for (int off = 128; off; off >>= 1) { if (tid < off) red[tid] += red[tid+off]; __syncthreads(); }
  float gsum = red[0];
#pragma unroll
  for (int d2 = 0; d2 < 32; ++d2) {
    float v = acc[d2];
    for (int off = 16; off; off >>= 1) v += __shfl_xor(v, off, 32);
    if ((tid & 31) == 0) atomicAdd(&sO[d2], v);
  }
  __syncthreads();
  if (tid < 32) O[((size_t)(b*nq+qi))*DD + h*32 + tid] = sO[tid] / gsum;
}

// ---------------------------------------------------------------------------
// Mamba selective scan, fused with D-skip and z-gate. dbl has row stride 64.
// ---------------------------------------------------------------------------
__global__ __launch_bounds__(128)
void scan_kernel(const float* __restrict__ dt, const float* __restrict__ xc,
                 const float* __restrict__ dbl, const float* __restrict__ z,
                 const float* __restrict__ A_log, const float* __restrict__ Dp,
                 float* __restrict__ y)
{
  int b = blockIdx.x;
  int c = blockIdx.y * 128 + threadIdx.x;
  float a[16], st[16];
#pragma unroll
  for (int s = 0; s < 16; ++s) { a[s] = -__expf(A_log[c*16+s]); st[s] = 0.f; }
  float dpv = Dp[c];
  for (int l = 0; l < GG; ++l) {
    size_t row = (size_t)b*GG + l;
    float dtv = dt[row*DI + c];
    float xv  = xc[row*DI + c];
    const float* dr = dbl + row*64;
    float du = dtv * xv, yv = 0.f;
#pragma unroll
    for (int s = 0; s < 16; ++s) {
      float dA = __expf(dtv * a[s]);
      st[s] = st[s]*dA + du*dr[16+s];
      yv += st[s]*dr[32+s];
    }
    float zv = z[row*DI + c];
    y[row*DI + c] = (yv + xv*dpv) * (zv/(1.f+__expf(-zv)));
  }
}

// Depthwise causal conv1d (D_CONV=4) + bias + silu
__global__ void conv_silu_kernel(const float* __restrict__ xp, const float* __restrict__ w,
                                 const float* __restrict__ bias, float* __restrict__ out)
{
  size_t i = (size_t)blockIdx.x*256 + threadIdx.x;
  if (i >= (size_t)NPTS*DI) return;
  int c = (int)(i % DI);
  size_t n = i / DI;
  int l = (int)(n % GG); int b = (int)(n / GG);
  float acc = bias[c];
#pragma unroll
  for (int t = 0; t < 4; ++t) {
    int l2 = l - 3 + t;
    if (l2 >= 0) acc += xp[((size_t)b*GG + l2)*DI + c] * w[c*4 + t];
  }
  out[i] = silu_f(acc);
}

// Elementwise helpers --------------------------------------------------------
__global__ void build_in_kernel(const float* __restrict__ coord, const float* __restrict__ feat,
                                const int* __restrict__ soi, float* __restrict__ out)
{
  size_t i = (size_t)blockIdx.x*256 + threadIdx.x;
  if (i >= (size_t)NPTS*32) return;
  int c = (int)(i % 32); size_t n = i / 32;
  float v = 0.f;
  if (c < 3)       v = coord[n*3 + c];
  else if (c < 9)  v = feat[n*6 + (c-3)];
  else if (c == 9) v = (float)soi[n / GG];
  out[i] = v;
}
// zero-padded weight copy: dst[drows,dcols] from src[srows,scols]
__global__ void pad_w_kernel(const float* __restrict__ src, float* __restrict__ dst,
                             int srows, int scols, int drows, int dcols)
{
  size_t i = (size_t)blockIdx.x*256 + threadIdx.x;
  if (i >= (size_t)drows*dcols) return;
  int r = (int)(i / dcols), c = (int)(i % dcols);
  dst[i] = (r < srows && c < scols) ? src[(size_t)r*scols + c] : 0.f;
}
__global__ void add_kernel(const float* a, const float* b, float* o, size_t n)
{ size_t i = (size_t)blockIdx.x*256 + threadIdx.x; if (i < n) o[i] = a[i] + b[i]; }
__global__ void axpy_kernel(float* a, const float* b, size_t n)
{ size_t i = (size_t)blockIdx.x*256 + threadIdx.x; if (i < n) a[i] += b[i]; }
__global__ void scale_kernel(const float* a, float s, float* o, size_t n)
{ size_t i = (size_t)blockIdx.x*256 + threadIdx.x; if (i < n) o[i] = a[i]*s; }
__global__ void fill_kernel(float* a, size_t n)
{ size_t i = (size_t)blockIdx.x*256 + threadIdx.x; if (i < n) a[i] = 0.f; }
__global__ void copy_kernel(const float* a, float* o, size_t n)
{ size_t i = (size_t)blockIdx.x*256 + threadIdx.x; if (i < n) o[i] = a[i]; }
__global__ void gather_cpe_kernel(const float* __restrict__ h, const int* __restrict__ nb,
                                  float* __restrict__ ag)
{
  size_t i = (size_t)blockIdx.x*256 + threadIdx.x;
  if (i >= (size_t)NPTS*2048) return;
  int d = (int)(i % DD); int k = (int)((i / DD) % 8); size_t n = i / 2048;
  ag[i] = h[(size_t)nb[n*8 + k]*DD + d];
}
__global__ void scatter_set_kernel(const float* __restrict__ src, const int* __restrict__ ord,
                                   float* __restrict__ dst)
{ int j = blockIdx.x, d = threadIdx.x; dst[(size_t)ord[j]*DD + d] = src[(size_t)j*DD + d]; }
__global__ void scatter_add_kernel(const float* __restrict__ src, const int* __restrict__ inv,
                                   float* __restrict__ dst)
{ int j = blockIdx.x, d = threadIdx.x; dst[(size_t)inv[j]*DD + d] += src[(size_t)j*DD + d]; }

// ---------------------------------------------------------------------------
static inline void gemm(hipStream_t s, const float* A, int lda, const float* B, int ldb,
                        const float* bias, float* C, int ldc, int M, int Ncol, int K, int act)
{
  dim3 g(Ncol/64, M/128);
  gemm_kernel<<<g, 256, 0, s>>>(A, lda, B, ldb, bias, C, ldc, K, act);
}
static inline size_t cdiv(size_t a, size_t b) { return (a + b - 1)/b; }

extern "C" void kernel_launch(void* const* d_in, const int* in_sizes, int n_in,
                              void* d_out, int out_size, void* d_ws, size_t ws_size,
                              hipStream_t stream)
{
  // ---- input map: jax pytree leaves, dict keys alphabetically sorted -------
  int pi = 0;
  const float* coord = (const float*)d_in[pi++];
  const float* feat  = (const float*)d_in[pi++];
  const int*   nbidx = (const int*)  d_in[pi++];
  struct Blk { const float *A_log,*Dp,*conv_b,*conv_w,*dt_b,*dt_w,*in_proj,*ln_b,*ln_w,*out_proj,*x_proj; } blk[2];
  for (int d = 0; d < 2; ++d) {
    blk[d].A_log   = (const float*)d_in[pi++];
    blk[d].Dp      = (const float*)d_in[pi++];
    blk[d].conv_b  = (const float*)d_in[pi++];
    blk[d].conv_w  = (const float*)d_in[pi++];
    blk[d].dt_b    = (const float*)d_in[pi++];
    blk[d].dt_w    = (const float*)d_in[pi++];
    blk[d].in_proj = (const float*)d_in[pi++];
    blk[d].ln_b    = (const float*)d_in[pi++];
    blk[d].ln_w    = (const float*)d_in[pi++];
    blk[d].out_proj= (const float*)d_in[pi++];
    blk[d].x_proj  = (const float*)d_in[pi++];
  }
  const float *c2q_x_wk=(const float*)d_in[pi++], *c2q_x_wo=(const float*)d_in[pi++],
              *c2q_x_wq=(const float*)d_in[pi++], *c2q_x_wv=(const float*)d_in[pi++];
  const float *c2q_fb1=(const float*)d_in[pi++], *c2q_fb2=(const float*)d_in[pi++],
              *c2q_fw1=(const float*)d_in[pi++], *c2q_fw2=(const float*)d_in[pi++];
  const float *c2q_l1b=(const float*)d_in[pi++], *c2q_l1w=(const float*)d_in[pi++],
              *c2q_l2b=(const float*)d_in[pi++], *c2q_l2w=(const float*)d_in[pi++],
              *c2q_l3b=(const float*)d_in[pi++], *c2q_l3w=(const float*)d_in[pi++];
  const float *c2q_s_wk=(const float*)d_in[pi++], *c2q_s_wo=(const float*)d_in[pi++],
              *c2q_s_wq=(const float*)d_in[pi++], *c2q_s_wv=(const float*)d_in[pi++];
  const float *cpe_b=(const float*)d_in[pi++], *cpe_fc_b=(const float*)d_in[pi++],
              *cpe_fc_w=(const float*)d_in[pi++], *cpe_ln_b=(const float*)d_in[pi++],
              *cpe_ln_w=(const float*)d_in[pi++], *cpe_w=(const float*)d_in[pi++];
  const float *enc_b1=(const float*)d_in[pi++], *enc_b2=(const float*)d_in[pi++],
              *enc_w1=(const float*)d_in[pi++], *enc_w2=(const float*)d_in[pi++];
  const float *norm_f_b=(const float*)d_in[pi++], *norm_f_w=(const float*)d_in[pi++];
  const float *q2c_x_wk=(const float*)d_in[pi++], *q2c_x_wo=(const float*)d_in[pi++],
              *q2c_x_wq=(const float*)d_in[pi++], *q2c_x_wv=(const float*)d_in[pi++];
  const float *q2c_fb1=(const float*)d_in[pi++], *q2c_fb2=(const float*)d_in[pi++],
              *q2c_fw1=(const float*)d_in[pi++], *q2c_fw2=(const float*)d_in[pi++];
  const float *q2c_l1b=(const float*)d_in[pi++], *q2c_l1w=(const float*)d_in[pi++],
              *q2c_l2b=(const float*)d_in[pi++], *q2c_l2w=(const float*)d_in[pi++];
  const float *query=(const float*)d_in[pi++], *query_pos=(const float*)d_in[pi++];
  const int *s_inverse=(const int*)d_in[pi++];
  const int *s_o_i=(const int*)d_in[pi++];
  const int *s_order=(const int*)d_in[pi++];
  (void)in_sizes; (void)n_in; (void)out_size; (void)ws_size;

  // ---- workspace arenas ----------------------------------------------------
  float* W = (float*)d_ws; size_t off = 0;
  auto alloc = [&](size_t nf){ float* p = W + off; off += nf; return p; };
  float* w_in10 = alloc((size_t)NPTS*32);     // encoder input, K padded 10->32
  float* w_h    = alloc((size_t)NPTS*DD);
  float* w_t0   = alloc((size_t)NPTS*1024);
  float* w_t1   = alloc((size_t)NPTS*DI);
  float* w_t2   = alloc((size_t)NPTS*DI);
  float* w_t3   = alloc((size_t)NPTS*DI);
  float* w_t4   = alloc((size_t)NPTS*DI);
  float* w_dbl  = alloc((size_t)NPTS*64);     // dbl, N padded 48->64
  float* w_seq  = alloc((size_t)NPTS*DD);
  float* w_r    = alloc((size_t)NPTS*DD);
  float* w_ln   = alloc((size_t)NPTS*DD);
  float* w_m    = alloc((size_t)NPTS*DD);
  float* w_acc  = alloc((size_t)NPTS*DD);
  float* w_ag   = w_t0;              // [NPTS,2048] aliases t0..t2 span (free at CPE time)
  float* w_k    = w_t3;              // enc K proj (free before mamba)
  float* w_v    = w_t4;              // enc V proj
  float* w_q    = alloc((size_t)NQP*DD);      // query buffers padded to 256 rows
  float* w_qt   = alloc((size_t)NQP*DD);
  float* w_qq   = alloc((size_t)NQP*DD);
  float* w_qk   = alloc((size_t)NQP*DD);
  float* w_qv   = alloc((size_t)NQP*DD);
  float* w_qa   = alloc((size_t)NQP*DD);
  float* w_qb   = alloc((size_t)NQP*DD);
  float* w_qh   = alloc((size_t)NQP*1024);
  float* w_ew1p = alloc((size_t)32*512);      // enc_w1 padded 10x512 -> 32x512
  float* w_xpp[2] = { alloc((size_t)512*64), alloc((size_t)512*64) };   // x_proj -> 512x64
  float* w_dtp[2] = { alloc((size_t)32*512), alloc((size_t)32*512) };   // dt_w   -> 32x512
  float* outp   = (float*)d_out;

  const size_t ND = (size_t)NPTS*DD, NQD = (size_t)NQR*DD;

  // ---- 0. pad awkward weights so every GEMM is guard-free ------------------
  pad_w_kernel<<<cdiv((size_t)32*512,256),256,0,stream>>>(enc_w1, w_ew1p, 10,512, 32,512);
  for (int d = 0; d < 2; ++d) {
    pad_w_kernel<<<cdiv((size_t)512*64,256),256,0,stream>>>(blk[d].x_proj, w_xpp[d], 512,48, 512,64);
    pad_w_kernel<<<cdiv((size_t)32*512,256),256,0,stream>>>(blk[d].dt_w,  w_dtp[d], 16,512, 32,512);
  }

  // ---- 1. encoder ----------------------------------------------------------
  build_in_kernel<<<cdiv((size_t)NPTS*32,256),256,0,stream>>>(coord, feat, s_o_i, w_in10);
  gemm(stream, w_in10,32, w_ew1p,512, enc_b1, w_t0,512, NPTS,512,32, 1);     // gelu
  gemm(stream, w_t0,512, enc_w2,256, enc_b2, w_h,256, NPTS,256,512, 0);

  // ---- 2. cloud_cross_query (queries attend to cloud) ----------------------
  copy_kernel<<<cdiv(NQD,256),256,0,stream>>>(query, w_q, NQD);
  add_kernel <<<cdiv(NQD,256),256,0,stream>>>(w_q, query_pos, w_qt, NQD);
  gemm(stream, w_qt,256, c2q_x_wq,256, nullptr, w_qq,256, NQP,256,256, 0);
  gemm(stream, w_h,256,  c2q_x_wk,256, nullptr, w_k,256,  NPTS,256,256, 0);
  gemm(stream, w_h,256,  c2q_x_wv,256, nullptr, w_v,256,  NPTS,256,256, 0);
  attn_kernel<<<dim3(NQ,8,BS),256,0,stream>>>(w_qq, w_k, w_v, w_qa, NQ, GG);
  gemm(stream, w_qa,256, c2q_x_wo,256, nullptr, w_qb,256, NQP,256,256, 0);
  ln_kernel<<<NQR,256,0,stream>>>(w_q, w_qb, c2q_l1w, c2q_l1b, w_q);
  add_kernel<<<cdiv(NQD,256),256,0,stream>>>(w_q, query_pos, w_qt, NQD);
  gemm(stream, w_qt,256, c2q_s_wq,256, nullptr, w_qq,256, NQP,256,256, 0);
  gemm(stream, w_qt,256, c2q_s_wk,256, nullptr, w_qk,256, NQP,256,256, 0);
  gemm(stream, w_q,256,  c2q_s_wv,256, nullptr, w_qv,256, NQP,256,256, 0);
  attn_kernel<<<dim3(NQ,8,BS),256,0,stream>>>(w_qq, w_qk, w_qv, w_qa, NQ, NQ);
  gemm(stream, w_qa,256, c2q_s_wo,256, nullptr, w_qb,256, NQP,256,256, 0);
  ln_kernel<<<NQR,256,0,stream>>>(w_q, w_qb, c2q_l2w, c2q_l2b, w_q);
  gemm(stream, w_q,256,  c2q_fw1,1024, c2q_fb1, w_qh,1024, NQP,1024,256, 1);
  gemm(stream, w_qh,1024,c2q_fw2,256,  c2q_fb2, w_qb,256,  NQP,256,1024, 0);
  ln_kernel<<<NQR,256,0,stream>>>(w_q, w_qb, c2q_l3w, c2q_l3b, w_q);

  // ---- 3. CPE --------------------------------------------------------------
  gather_cpe_kernel<<<cdiv((size_t)NPTS*2048,256),256,0,stream>>>(w_h, nbidx, w_ag);
  gemm(stream, w_ag,2048, cpe_w,256, cpe_b, w_seq,256, NPTS,256,2048, 0);
  gemm(stream, w_seq,256, cpe_fc_w,256, cpe_fc_b, w_r,256, NPTS,256,256, 0);
  ln_kernel<<<NPTS,256,0,stream>>>(w_r, nullptr, cpe_ln_w, cpe_ln_b, w_ln);
  axpy_kernel<<<cdiv(ND,256),256,0,stream>>>(w_h, w_ln, ND);

  // ---- 4. multi-order serialized Mamba mixer -------------------------------
  fill_kernel<<<cdiv(ND,256),256,0,stream>>>(w_acc, ND);
  for (int i = 0; i < 3; ++i) {
    scatter_set_kernel<<<NPTS,256,0,stream>>>(w_h, s_order + (size_t)i*NPTS, w_seq);
    copy_kernel<<<cdiv(ND,256),256,0,stream>>>(w_seq, w_r, ND);
    for (int d = 0; d < 2; ++d) {
      if (d) axpy_kernel<<<cdiv(ND,256),256,0,stream>>>(w_r, w_m, ND);
      ln_kernel<<<NPTS,256,0,stream>>>(w_r, nullptr, blk[d].ln_w, blk[d].ln_b, w_ln);
      gemm(stream, w_ln,256, blk[d].in_proj,     1024, nullptr, w_t1,512, NPTS,512,256, 0); // xp
      gemm(stream, w_ln,256, blk[d].in_proj+512, 1024, nullptr, w_t2,512, NPTS,512,256, 0); // z
      conv_silu_kernel<<<cdiv((size_t)NPTS*DI,256),256,0,stream>>>(w_t1, blk[d].conv_w, blk[d].conv_b, w_t3);
      gemm(stream, w_t3,512, w_xpp[d],64, nullptr, w_dbl,64, NPTS,64,512, 0);               // dbl (padded)
      gemm(stream, w_dbl,64, w_dtp[d],512, blk[d].dt_b, w_t4,512, NPTS,512,32, 3);          // softplus
      scan_kernel<<<dim3(BS,4),128,0,stream>>>(w_t4, w_t3, w_dbl, w_t2,
                                               blk[d].A_log, blk[d].Dp, w_t1);
      gemm(stream, w_t1,512, blk[d].out_proj,256, nullptr, w_m,256, NPTS,256,512, 0);
    }
    ln_kernel<<<NPTS,256,0,stream>>>(w_m, w_r, norm_f_w, norm_f_b, w_ln);
    scatter_add_kernel<<<NPTS,256,0,stream>>>(w_ln, s_inverse + (size_t)i*NPTS, w_acc);
  }
  scale_kernel<<<cdiv(ND,256),256,0,stream>>>(w_acc, 1.f/3.f, w_h, ND);

  // ---- 5. query_cross_cloud (cloud attends to queries) ---------------------
  gemm(stream, w_h,256, q2c_x_wq,256, nullptr, w_seq,256, NPTS,256,256, 0);
  gemm(stream, w_q,256, q2c_x_wk,256, nullptr, w_qk,256,  NQP,256,256, 0);
  gemm(stream, w_q,256, q2c_x_wv,256, nullptr, w_qv,256,  NQP,256,256, 0);
  attn_kernel<<<dim3(GG,8,BS),256,0,stream>>>(w_seq, w_qk, w_qv, w_r, GG, NQ);
  gemm(stream, w_r,256, q2c_x_wo,256, nullptr, w_m,256, NPTS,256,256, 0);
  ln_kernel<<<NPTS,256,0,stream>>>(w_h, w_m, q2c_l1w, q2c_l1b, w_h);
  gemm(stream, w_h,256,  q2c_fw1,1024, q2c_fb1, w_t0,1024, NPTS,1024,256, 1);
  gemm(stream, w_t0,1024,q2c_fw2,256,  q2c_fb2, w_m,256,   NPTS,256,1024, 0);
  ln_kernel<<<NPTS,256,0,stream>>>(w_h, w_m, q2c_l2w, q2c_l2b, outp);
}
